// STHD_SpGAT_75814762709172
// MI455X (gfx1250) — matrix operand order repaired and checked by reading, not compiled, not run on hardware
//
#include <hip/hip_runtime.h>
#include <hip/hip_bf16.h>
#include <math.h>

// ---------------------------------------------------------------------------
// Problem constants (match reference)
// ---------------------------------------------------------------------------
constexpr int N   = 50000;   // cells
constexpr int C   = 30;      // classes
constexpr int G   = 1000;    // genes
constexpr int E   = 800000;  // edges
constexpr int OUT = 8;       // GATv2 out channels
constexpr int GP  = 1024;    // G padded to multiple of 32 (WMMA K chunk)
constexpr int KCH = 32;      // K per v_wmma_f32_16x16x32_bf16

typedef __attribute__((ext_vector_type(16))) __bf16 v16bf;
typedef __attribute__((ext_vector_type(8)))  __bf16 v8bf;
typedef __attribute__((ext_vector_type(8)))  float  v8f;

// ---------------------------------------------------------------------------
// Workspace layout (bytes)
// ---------------------------------------------------------------------------
constexpr size_t OFF_BINV  = 0;                              // [32][1024] bf16  (64 KB)
constexpr size_t OFF_BMU   = OFF_BINV + 32 * GP * 2;         // [32][1024] bf16  (64 KB)
constexpr size_t OFF_BLR   = OFF_BMU  + 32 * GP * 2;         // [16][1024] bf16  (32 KB)
constexpr size_t OFF_D     = OFF_BLR  + 16 * GP * 2;         // [32] f32
constexpr size_t OFF_XL    = OFF_D    + 32 * 4 + 128;        // [N][8] f32
constexpr size_t OFF_XR    = OFF_XL   + (size_t)N * 8 * 4;   // [N][8] f32
constexpr size_t OFF_EV    = OFF_XR   + (size_t)N * 8 * 4;   // [E] f32 raw scores
constexpr size_t OFF_EEXP  = OFF_EV   + (size_t)E * 4;       // [E] f32 exp
constexpr size_t OFF_EMAX  = OFF_EEXP + (size_t)E * 4;       // [N] uint (order-coded max)
constexpr size_t OFF_DEN   = OFF_EMAX + (size_t)N * 4;       // [N] f32 denom

// ---------------------------------------------------------------------------
// Helpers
// ---------------------------------------------------------------------------
__device__ __forceinline__ unsigned short bf16_of(float x) {
    union { __bf16 b; unsigned short u; } c;
    c.b = (__bf16)x;
    return c.u;
}
// order-preserving float<->uint encoding for atomicMax-based segment max
__device__ __forceinline__ unsigned f2ord(float f) {
    unsigned u = __float_as_uint(f);
    return (u & 0x80000000u) ? ~u : (u | 0x80000000u);
}
__device__ __forceinline__ float ord2f(unsigned o) {
    unsigned u = (o & 0x80000000u) ? (o ^ 0x80000000u) : ~o;
    return __uint_as_float(u);
}

// Assemble a 16-wide bf16 A-fragment from two 8-wide f32 groups.
// convertvector lowers to paired v_cvt_pk_bf16_f32; shufflevector is a concat.
__device__ __forceinline__ v16bf make_afrag(v8f lo, v8f hi) {
    v8bf blo = __builtin_convertvector(lo, v8bf);
    v8bf bhi = __builtin_convertvector(hi, v8bf);
    return __builtin_shufflevector(blo, bhi,
                                   0, 1, 2, 3, 4, 5, 6, 7,
                                   8, 9, 10, 11, 12, 13, 14, 15);
}

// B-fragment loader: lane holds 16 consecutive K (bf16) of one column.
// Memory layout is class-major [row][k] so this is one contiguous 32B read.
__device__ __forceinline__ v16bf load_bfrag(const unsigned short* base, int idx) {
    union { uint4 q[2]; v16bf v; } f;
    const uint4* p = reinterpret_cast<const uint4*>(base + idx);
    f.q[0] = p[0];
    f.q[1] = p[1];
    return f.v;
}

// 8-float vector load (two aligned b128)
__device__ __forceinline__ v8f load_f8(const float* p) {
    union { float4 q[2]; v8f v; } f;
    const float4* q = reinterpret_cast<const float4*>(p);
    f.q[0] = q[0];
    f.q[1] = q[1];
    return f.v;
}

// ---------------------------------------------------------------------------
// Kernel 0: build bf16 B-tiles (zero padded to [32|16][1024]) + D[c]
//   blocks 0..31  : class rows  -> Binv = 1/Var, Bmu = Mu/Var, D[c]=sum Mu^2/Var
//   blocks 32..47 : lin rows    -> Blr rows 0-7 = Wl, rows 8-15 = Wr
// ---------------------------------------------------------------------------
__global__ __launch_bounds__(256) void k_prep(const float* __restrict__ Mu,
                                              const float* __restrict__ Var,
                                              const float* __restrict__ Wl,
                                              const float* __restrict__ Wr,
                                              unsigned short* __restrict__ Binv,
                                              unsigned short* __restrict__ Bmu,
                                              unsigned short* __restrict__ Blr,
                                              float* __restrict__ Dv) {
    __shared__ float red[256];
    const int r = blockIdx.x;
    const int t = threadIdx.x;
    if (r < 32) {
        const int c = r;
        float dsum = 0.0f;
        for (int k = t; k < GP; k += 256) {
            float inv = 0.0f, mu = 0.0f;
            if (c < C && k < G) {
                inv = 1.0f / Var[c * G + k];
                mu  = Mu[c * G + k];
            }
            Binv[c * GP + k] = bf16_of(inv);
            Bmu [c * GP + k] = bf16_of(mu * inv);
            dsum += mu * mu * inv;
        }
        red[t] = dsum;
        __syncthreads();
        for (int s = 128; s > 0; s >>= 1) {
            if (t < s) red[t] += red[t + s];
            __syncthreads();
        }
        if (t == 0) Dv[c] = red[0];
    } else {
        const int j = r - 32;                        // 0..15
        const float* src = (j < OUT) ? (Wl + (size_t)j * G) : (Wr + (size_t)(j - OUT) * G);
        for (int k = t; k < GP; k += 256) {
            float v = (k < G) ? src[k] : 0.0f;
            Blr[j * GP + k] = bf16_of(v);
        }
    }
}

// ---------------------------------------------------------------------------
// Kernel 1: init accumulators / segment buffers (re-run every call: graph safe)
// ---------------------------------------------------------------------------
__global__ __launch_bounds__(256) void k_init(unsigned* __restrict__ emax,
                                              float* __restrict__ denom,
                                              float* __restrict__ out) {
    int i = blockIdx.x * 256 + threadIdx.x;
    if (i < N) {
        emax[i]  = f2ord(-3.402823466e38f);
        denom[i] = 0.0f;
    }
    if (i < 2) out[i] = 0.0f;
}

// ---------------------------------------------------------------------------
// Kernel 2: fused triple-GEMM over one pass of X (bf16 WMMA, f32 accum)
//   per wave: 16 rows of X; 5 wmma per K=32 chunk:
//     accA = (X*X) @ invVar^T   (2 column tiles: classes 0-15 / 16-31)
//     accB =  X    @ (Mu*invVar)^T
//     accL =  X    @ [Wl|Wr]^T  (16 columns)
//   epilogue: softmax(W row), Fmat, ll_prot partial, P out, x_l/x_r out
// ---------------------------------------------------------------------------
__global__ __launch_bounds__(256) void k_fused_gemm(
        const float* __restrict__ X,
        const float* __restrict__ W,
        const float* __restrict__ S,
        const float* __restrict__ bl,
        const float* __restrict__ br,
        const unsigned short* __restrict__ Binv,
        const unsigned short* __restrict__ Bmu,
        const unsigned short* __restrict__ Blr,
        const float* __restrict__ Dv,
        float* __restrict__ out,          // [0]=ll, [1]=ce, [2..]=P
        float* __restrict__ Xl,
        float* __restrict__ Xr) {
    __shared__ float smem[8 * 1280];      // per wave: A 16x32 | B 16x32 | L 16x16

    const int lane = threadIdx.x & 31;
    const int wv   = threadIdx.x >> 5;
    const int m    = lane & 15;           // row within tile (A layout)
    const int hi   = lane >> 4;           // K-half selector (A & B layouts)
    const int rowBase = blockIdx.x * 128 + wv * 16;
    const int row  = rowBase + m;
    const int rowc = (row < N) ? row : (N - 1);
    const float* xrow = X + (size_t)rowc * G;

    v8f accA0 = {}, accA1 = {}, accB0 = {}, accB1 = {}, accL = {};

    // ---- main K loop: chunks of 32, 31 full chunks cover K=0..991 ----
    for (int kc = 0; kc < 992; kc += KCH) {
        const int o0 = kc + 16 * hi;      // lanes' first 8 K values
        const int o1 = kc + 16 * hi + 8;  // lanes' second 8 K values
        if (kc + KCH < 992) __builtin_prefetch(xrow + kc + KCH + 8 * hi, 0, 0);

        v8f x0 = load_f8(xrow + o0);
        v8f x1 = load_f8(xrow + o1);

        v16bf a1 = make_afrag(x0, x1);            // X
        v16bf a2 = make_afrag(x0 * x0, x1 * x1);  // X^2 (v_pk_mul_f32)

        const int kk = kc + 16 * hi;      // B layout: high lanes hold K=16..31
        v16bf bi0 = load_bfrag(Binv, ( 0 + m) * GP + kk);
        v16bf bi1 = load_bfrag(Binv, (16 + m) * GP + kk);
        v16bf bm0 = load_bfrag(Bmu,  ( 0 + m) * GP + kk);
        v16bf bm1 = load_bfrag(Bmu,  (16 + m) * GP + kk);
        v16bf blr = load_bfrag(Blr,  m * GP + kk);

        accA0 = __builtin_amdgcn_wmma_f32_16x16x32_bf16(false, a2, false, bi0, (short)0, accA0, false, false);
        accA1 = __builtin_amdgcn_wmma_f32_16x16x32_bf16(false, a2, false, bi1, (short)0, accA1, false, false);
        accB0 = __builtin_amdgcn_wmma_f32_16x16x32_bf16(false, a1, false, bm0, (short)0, accB0, false, false);
        accB1 = __builtin_amdgcn_wmma_f32_16x16x32_bf16(false, a1, false, bm1, (short)0, accB1, false, false);
        accL  = __builtin_amdgcn_wmma_f32_16x16x32_bf16(false, a1, false, blr, (short)0, accL,  false, false);
    }

    // ---- tail chunk kc=992 (K 992..1023, only 992..999 valid; B zero-padded) ----
    {
        v8f x0;
        #pragma unroll
        for (int i = 0; i < 8; ++i) {
            int k = 992 + 16 * hi + i;
            x0[i] = (k < G) ? xrow[k] : 0.0f;
        }
        v8f zz = {};
        v16bf a1 = make_afrag(x0, zz);
        v16bf a2 = make_afrag(x0 * x0, zz);

        const int kk = 992 + 16 * hi;
        v16bf bi0 = load_bfrag(Binv, ( 0 + m) * GP + kk);
        v16bf bi1 = load_bfrag(Binv, (16 + m) * GP + kk);
        v16bf bm0 = load_bfrag(Bmu,  ( 0 + m) * GP + kk);
        v16bf bm1 = load_bfrag(Bmu,  (16 + m) * GP + kk);
        v16bf blr = load_bfrag(Blr,  m * GP + kk);

        accA0 = __builtin_amdgcn_wmma_f32_16x16x32_bf16(false, a2, false, bi0, (short)0, accA0, false, false);
        accA1 = __builtin_amdgcn_wmma_f32_16x16x32_bf16(false, a2, false, bi1, (short)0, accA1, false, false);
        accB0 = __builtin_amdgcn_wmma_f32_16x16x32_bf16(false, a1, false, bm0, (short)0, accB0, false, false);
        accB1 = __builtin_amdgcn_wmma_f32_16x16x32_bf16(false, a1, false, bm1, (short)0, accB1, false, false);
        accL  = __builtin_amdgcn_wmma_f32_16x16x32_bf16(false, a1, false, blr, (short)0, accL,  false, false);
    }

    // ---- spill accumulators: C/D layout is row = v + 8*hi, col = lane&15 ----
    float* sw   = smem + wv * 1280;
    float* ldsA = sw;            // [16][32]
    float* ldsB = sw + 512;      // [16][32]
    float* ldsL = sw + 1024;     // [16][16]
    #pragma unroll
    for (int v = 0; v < 8; ++v) {
        int rr = v + 8 * hi;
        ldsA[rr * 32 +      m] = accA0[v];
        ldsA[rr * 32 + 16 + m] = accA1[v];
        ldsB[rr * 32 +      m] = accB0[v];
        ldsB[rr * 32 + 16 + m] = accB1[v];
        ldsL[rr * 16 +      m] = accL[v];
    }
    __syncthreads();

    // ---- per-row epilogue on lanes 0..15 ----
    if (lane < 16 && row < N) {
        const float s  = S[row];
        const float* wrow = W + (size_t)row * C;

        float mx = -3.402823466e38f;
        #pragma unroll
        for (int c = 0; c < C; ++c) mx = fmaxf(mx, wrow[c]);
        float pv[C];
        float sum = 0.0f;
        #pragma unroll
        for (int c = 0; c < C; ++c) { pv[c] = __expf(wrow[c] - mx); sum += pv[c]; }
        const float inv = 1.0f / sum;

        float ll = 0.0f;
        float* outP = out + 2 + (size_t)row * C;
        #pragma unroll
        for (int c = 0; c < C; ++c) {
            float p = pv[c] * inv;
            float a = ldsA[lane * 32 + c];
            float b = ldsB[lane * 32 + c];
            float f = -0.5f * (a - 2.0f * s * b + s * s * Dv[c]);
            ll += p * f;
            outP[c] = p;
        }
        atomicAdd(out, ll * (1.0f / (float)N));

        #pragma unroll
        for (int j = 0; j < OUT; ++j) {
            Xl[(size_t)row * OUT + j] = ldsL[lane * 16 + j]       + bl[j];
            Xr[(size_t)row * OUT + j] = ldsL[lane * 16 + OUT + j] + br[j];
        }
    }
}

// ---------------------------------------------------------------------------
// Kernel 3: edge scores + segment max (order-coded uint atomicMax)
// ---------------------------------------------------------------------------
__global__ __launch_bounds__(256) void k_edge1(const long long* __restrict__ ei,
                                               const float* __restrict__ Xl,
                                               const float* __restrict__ Xr,
                                               const float* __restrict__ att,
                                               float* __restrict__ ev,
                                               unsigned* __restrict__ emax) {
    int idx = blockIdx.x * 256 + threadIdx.x;
    if (idx >= E) return;
    int s = (int)ei[idx];
    int d = (int)ei[E + idx];
    float4 l0 = *reinterpret_cast<const float4*>(Xl + (size_t)s * OUT);
    float4 l1 = *reinterpret_cast<const float4*>(Xl + (size_t)s * OUT + 4);
    float4 r0 = *reinterpret_cast<const float4*>(Xr + (size_t)d * OUT);
    float4 r1 = *reinterpret_cast<const float4*>(Xr + (size_t)d * OUT + 4);
    float mj[OUT] = { l0.x + r0.x, l0.y + r0.y, l0.z + r0.z, l0.w + r0.w,
                      l1.x + r1.x, l1.y + r1.y, l1.z + r1.z, l1.w + r1.w };
    float e = 0.0f;
    #pragma unroll
    for (int j = 0; j < OUT; ++j) {
        float v = mj[j];
        v = (v > 0.0f) ? v : 0.2f * v;     // leaky_relu(0.2)
        e += att[j] * v;
    }
    ev[idx] = e;
    atomicMax(emax + d, f2ord(e));
}

// ---------------------------------------------------------------------------
// Kernel 4: exp(e - max) + segment sum
// ---------------------------------------------------------------------------
__global__ __launch_bounds__(256) void k_edge2(const long long* __restrict__ ei,
                                               const float* __restrict__ ev,
                                               const unsigned* __restrict__ emax,
                                               float* __restrict__ eexp,
                                               float* __restrict__ denom) {
    int idx = blockIdx.x * 256 + threadIdx.x;
    if (idx >= E) return;
    int d = (int)ei[E + idx];
    float ex = __expf(ev[idx] - ord2f(emax[d]));
    eexp[idx] = ex;
    atomicAdd(denom + d, ex);
}

// ---------------------------------------------------------------------------
// Kernel 5: CE reduction (wave-shuffle reduce, one atomic per wave)
// ---------------------------------------------------------------------------
__global__ __launch_bounds__(256) void k_edge3(const long long* __restrict__ ei,
                                               const float* __restrict__ eexp,
                                               const float* __restrict__ denom,
                                               const float* __restrict__ out_ro, // P at +2
                                               float* __restrict__ out) {
    int idx = blockIdx.x * 256 + threadIdx.x;
    float contrib = 0.0f;
    if (idx < E) {
        int s = (int)ei[idx];
        int d = (int)ei[E + idx];
        float alpha = eexp[idx] / (denom[d] + 1e-16f);
        const float* Ps = out_ro + 2 + (size_t)s * C;
        const float* Pd = out_ro + 2 + (size_t)d * C;
        float t = 0.0f;
        #pragma unroll
        for (int c = 0; c < C; ++c) t += Ps[c] * __logf(Pd[c] + 1e-8f);
        contrib = -alpha * t * (1.0f / (float)N);
    }
    #pragma unroll
    for (int off = 16; off > 0; off >>= 1)
        contrib += __shfl_xor(contrib, off, 32);
    if ((threadIdx.x & 31) == 0) atomicAdd(out + 1, contrib);
}

// ---------------------------------------------------------------------------
// Launcher
// ---------------------------------------------------------------------------
extern "C" void kernel_launch(void* const* d_in, const int* in_sizes, int n_in,
                              void* d_out, int out_size, void* d_ws, size_t ws_size,
                              hipStream_t stream) {
    const float*     X   = (const float*)d_in[0];
    const float*     Mu  = (const float*)d_in[1];
    const float*     Var = (const float*)d_in[2];
    const long long* ei  = (const long long*)d_in[3];   // int64 edge_index
    const float*     W   = (const float*)d_in[4];
    const float*     S   = (const float*)d_in[5];
    const float*     Wl  = (const float*)d_in[6];
    const float*     bl  = (const float*)d_in[7];
    const float*     Wr  = (const float*)d_in[8];
    const float*     br  = (const float*)d_in[9];
    const float*     att = (const float*)d_in[10];
    float* out = (float*)d_out;

    char* ws = (char*)d_ws;
    unsigned short* Binv = (unsigned short*)(ws + OFF_BINV);
    unsigned short* Bmu  = (unsigned short*)(ws + OFF_BMU);
    unsigned short* Blr  = (unsigned short*)(ws + OFF_BLR);
    float*    Dv    = (float*)(ws + OFF_D);
    float*    Xl    = (float*)(ws + OFF_XL);
    float*    Xr    = (float*)(ws + OFF_XR);
    float*    ev    = (float*)(ws + OFF_EV);
    float*    eexp  = (float*)(ws + OFF_EEXP);
    unsigned* emax  = (unsigned*)(ws + OFF_EMAX);
    float*    denom = (float*)(ws + OFF_DEN);

    k_prep<<<48, 256, 0, stream>>>(Mu, Var, Wl, Wr, Binv, Bmu, Blr, Dv);
    k_init<<<(N + 255) / 256, 256, 0, stream>>>(emax, denom, out);
    k_fused_gemm<<<(N + 127) / 128, 256, 0, stream>>>(X, W, S, bl, br,
                                                      Binv, Bmu, Blr, Dv,
                                                      out, Xl, Xr);
    k_edge1<<<(E + 255) / 256, 256, 0, stream>>>(ei, Xl, Xr, att, ev, emax);
    k_edge2<<<(E + 255) / 256, 256, 0, stream>>>(ei, ev, emax, eexp, denom);
    k_edge3<<<(E + 255) / 256, 256, 0, stream>>>(ei, eexp, denom, out, out);
}
